// MambaTransformerLayer_25580825215308
// MI455X (gfx1250) — matrix operand order repaired and checked by reading, not compiled, not run on hardware
//
#include <hip/hip_runtime.h>
#include <hip/hip_bf16.h>

// ---------------------------------------------------------------------------
// MI455X (gfx1250) Mamba-Transformer layer.
//  - Dense projections: v_wmma_f32_16x16x32_bf16, f32 accumulate, fused
//    bias/SiLU/GELU/residual epilogues, software-pipelined K loop
//    (register double-buffering so WMMAs overlap in-flight loads).
//  - SSM scan: chunked parallel scan (local scans + A^32 combine + fixup).
// ---------------------------------------------------------------------------

typedef __attribute__((ext_vector_type(16))) unsigned short v16u;
typedef __attribute__((ext_vector_type(8)))  unsigned short v8u;
typedef __attribute__((ext_vector_type(16))) __bf16        v16bf;
typedef __attribute__((ext_vector_type(8)))  float         v8f;

#define BATCH 4
#define SEQ   2048
#define DMODEL 1024
#define DSTATE 64
#define DFF    2048
#define MROWS (BATCH * SEQ)          // 8192
#define CHUNK 32
#define NCHUNK (SEQ / CHUNK)         // 64

enum { EPI_NONE = 0, EPI_SILU = 1, EPI_GELU = 2, EPI_RESID = 3 };

__device__ __forceinline__ unsigned short f2bf(float f) {
  unsigned int x = __builtin_bit_cast(unsigned int, f);
  unsigned int r;
  if ((x & 0x7F800000u) == 0x7F800000u) {          // inf / nan
    r = x;
    if (x & 0x007FFFFFu) r |= 0x00400000u;          // quiet the nan
  } else {
    r = x + (0x7FFFu + ((x >> 16) & 1u));           // round-to-nearest-even
  }
  return (unsigned short)(r >> 16);
}

__device__ __forceinline__ float silu_f(float v) {
  return v * (1.0f / (1.0f + __expf(-v)));
}
__device__ __forceinline__ float gelu_f(float v) {
  return 0.5f * v * (1.0f + erff(v * 0.70710678118654752f));
}

// B-fragment (32x16 K-major): 16 contiguous bf16 per lane.
__device__ __forceinline__ v16bf load_bfrag(const unsigned short* p) {
  return __builtin_bit_cast(v16bf, *(const v16u*)p);
}
// A-fragment (16x32): per lane, 16 contiguous K (k = 16*half + s).
__device__ __forceinline__ v16bf load_afrag(const unsigned short* p) {
  return __builtin_bit_cast(v16bf, *(const v16u*)p);
}

// ---------------------------------------------------------------------------
// bf16 WMMA GEMM:  C[M,N] = A[M,K] @ W[N,K]^T (+bias, +epilogue, +residual)
// Block: 256 threads = 8 waves; block tile 128(M) x 64*NT(N).
// Per wave: 4 M-tiles x NT N-tiles (64 x 16*NT output), K step 32,
// double-buffered fragments so loads for k+32 are in flight during the
// WMMAs for k (K must be a multiple of 64).
// ---------------------------------------------------------------------------
template <int EPI, bool OUT_BF16, int NT>
__global__ __launch_bounds__(256) void gemm_bf16_kernel(
    const unsigned short* __restrict__ A,   // [M,K] bf16 row-major
    const unsigned short* __restrict__ W,   // [N,K] bf16 row-major
    const float* __restrict__ bias,         // [N] or nullptr
    const float* __restrict__ resid,        // [M,N] or nullptr
    void* __restrict__ Cout, int M, int N, int K) {
  const int tid  = threadIdx.x;
  const int wid  = tid >> 5;
  const int lane = tid & 31;
  const int l16  = lane & 15;
  const int lh   = lane >> 4;

  const int tileN = blockIdx.x * (64 * NT) + (wid >> 1) * (16 * NT);
  const int tileM = blockIdx.y * 128 + (wid & 1) * 64;

  v8f acc[4][NT];
  const v8f zero8 = {0.f, 0.f, 0.f, 0.f, 0.f, 0.f, 0.f, 0.f};
#pragma unroll
  for (int i = 0; i < 4; ++i)
#pragma unroll
    for (int n = 0; n < NT; ++n) acc[i][n] = zero8;

  const unsigned short* wp[NT];
#pragma unroll
  for (int n = 0; n < NT; ++n)
    wp[n] = W + (size_t)(tileN + n * 16 + l16) * K + lh * 16;
  const unsigned short* arow = A + (size_t)(tileM + l16) * K + lh * 16;

  v16bf aF[2][4], bF[2][NT];
  // prime pipeline with kb = 0
#pragma unroll
  for (int n = 0; n < NT; ++n) bF[0][n] = load_bfrag(wp[n]);
#pragma unroll
  for (int i = 0; i < 4; ++i) aF[0][i] = load_afrag(arow + (size_t)(i * 16) * K);

  for (int kb = 0; kb < K; kb += 64) {
    // stage kb+32 while computing kb
    __builtin_prefetch(wp[0] + kb + 512, 0, 1);     // global_prefetch_b8
    __builtin_prefetch(arow + kb + 512, 0, 1);
#pragma unroll
    for (int n = 0; n < NT; ++n) bF[1][n] = load_bfrag(wp[n] + kb + 32);
#pragma unroll
    for (int i = 0; i < 4; ++i)
      aF[1][i] = load_afrag(arow + (size_t)(i * 16) * K + kb + 32);
#pragma unroll
    for (int i = 0; i < 4; ++i)
#pragma unroll
      for (int n = 0; n < NT; ++n)
        acc[i][n] = __builtin_amdgcn_wmma_f32_16x16x32_bf16(
            false, aF[0][i], false, bF[0][n], (short)0, acc[i][n], false, false);

    if (kb + 64 < K) {
      // stage kb+64 while computing kb+32
#pragma unroll
      for (int n = 0; n < NT; ++n) bF[0][n] = load_bfrag(wp[n] + kb + 64);
#pragma unroll
      for (int i = 0; i < 4; ++i)
        aF[0][i] = load_afrag(arow + (size_t)(i * 16) * K + kb + 64);
    }
#pragma unroll
    for (int i = 0; i < 4; ++i)
#pragma unroll
      for (int n = 0; n < NT; ++n)
        acc[i][n] = __builtin_amdgcn_wmma_f32_16x16x32_bf16(
            false, aF[1][i], false, bF[1][n], (short)0, acc[i][n], false, false);
  }

  float* cf = (float*)Cout;
  unsigned short* cb = (unsigned short*)Cout;
#pragma unroll
  for (int n = 0; n < NT; ++n) {
    const int col = tileN + n * 16 + l16;
    const float bv = (bias != nullptr) ? bias[col] : 0.0f;
#pragma unroll
    for (int i = 0; i < 4; ++i) {
#pragma unroll
      for (int r = 0; r < 8; ++r) {
        const int row = tileM + i * 16 + lh * 8 + r;
        float v = acc[i][n][r] + bv;
        if (EPI == EPI_SILU) v = silu_f(v);
        if (EPI == EPI_GELU) v = gelu_f(v);
        size_t idx = (size_t)row * N + col;
        if (EPI == EPI_RESID) v += resid[idx];
        if (OUT_BF16) cb[idx] = f2bf(v);
        else          cf[idx] = v;
      }
    }
  }
}

// ---------------------------------------------------------------------------
// LayerNorm over last dim (D=1024) -> bf16 output (feeds WMMA A-operand).
// ---------------------------------------------------------------------------
__global__ __launch_bounds__(256) void layernorm_bf16_kernel(
    const float* __restrict__ x, const float* __restrict__ g,
    const float* __restrict__ b, unsigned short* __restrict__ out, int D) {
  const int row = blockIdx.x;
  const float* xr = x + (size_t)row * D;
  float s = 0.f, ss = 0.f;
  for (int i = threadIdx.x; i < D; i += 256) {
    float v = xr[i];
    s += v;
    ss += v * v;
  }
  __shared__ float red0[256], red1[256];
  red0[threadIdx.x] = s;
  red1[threadIdx.x] = ss;
  __syncthreads();
  for (int off = 128; off > 0; off >>= 1) {
    if (threadIdx.x < off) {
      red0[threadIdx.x] += red0[threadIdx.x + off];
      red1[threadIdx.x] += red1[threadIdx.x + off];
    }
    __syncthreads();
  }
  const float mu = red0[0] / (float)D;
  const float var = red1[0] / (float)D - mu * mu;
  const float rs = rsqrtf(var + 1e-5f);
  for (int i = threadIdx.x; i < D; i += 256)
    out[(size_t)row * D + i] = f2bf((xr[i] - mu) * rs * g[i] + b[i]);
}

// ---------------------------------------------------------------------------
// Depthwise conv1d (k=3, pad=1, groups=D) + SiLU -> bf16
// ---------------------------------------------------------------------------
__global__ __launch_bounds__(256) void conv_silu_kernel(
    const float* __restrict__ x, const float* __restrict__ w,
    unsigned short* __restrict__ out, int S, int D, long long total) {
  long long idx = (long long)blockIdx.x * blockDim.x + threadIdx.x;
  if (idx >= total) return;
  const int d = (int)(idx % D);
  const long long sd = idx / D;
  const int s = (int)(sd % S);
  const long long b = sd / S;
  const float* xc = x + (b * S) * (size_t)D + d;
  const float w0 = w[d * 3 + 0], w1 = w[d * 3 + 1], w2 = w[d * 3 + 2];
  const float xm = (s > 0) ? xc[(size_t)(s - 1) * D] : 0.0f;
  const float x0 = xc[(size_t)s * D];
  const float xp = (s < S - 1) ? xc[(size_t)(s + 1) * D] : 0.0f;
  float y = w0 * xm + w1 * x0 + w2 * xp;
  out[idx] = f2bf(silu_f(y));
}

// ---------------------------------------------------------------------------
// Elementwise helpers
// ---------------------------------------------------------------------------
__global__ __launch_bounds__(256) void f32_to_bf16_kernel(
    const float* __restrict__ s, unsigned short* __restrict__ d, long long n) {
  long long i = (long long)blockIdx.x * blockDim.x + threadIdx.x;
  for (; i < n; i += (long long)gridDim.x * blockDim.x) d[i] = f2bf(s[i]);
}

__global__ __launch_bounds__(256) void gate_bf16_kernel(
    const float* __restrict__ xs, const float* __restrict__ v,
    unsigned short* __restrict__ g, long long n) {
  long long i = (long long)blockIdx.x * blockDim.x + threadIdx.x;
  if (i < n) g[i] = f2bf(xs[i] * v[i]);
}

// ---------------------------------------------------------------------------
// SSM chunked parallel scan.  h_t = A h_{t-1} + u_t (note (h A^T)_j = (A h)_j)
// ---------------------------------------------------------------------------

// A32 = A^32 via 5 squarings.  One block of 64 threads, row-per-thread.
__global__ __launch_bounds__(64) void apow32_kernel(
    const float* __restrict__ A, float* __restrict__ A32) {
  __shared__ float Ms[DSTATE * DSTATE];
  const int j = threadIdx.x;
  for (int c = 0; c < DSTATE; ++c) Ms[j * DSTATE + c] = A[j * DSTATE + c];
  __syncthreads();
  float row[DSTATE];
  for (int it = 0; it < 5; ++it) {
    for (int c = 0; c < DSTATE; ++c) {
      float acc = 0.f;
#pragma unroll 8
      for (int k = 0; k < DSTATE; ++k) acc += Ms[j * DSTATE + k] * Ms[k * DSTATE + c];
      row[c] = acc;
    }
    __syncthreads();
    for (int c = 0; c < DSTATE; ++c) Ms[j * DSTATE + c] = row[c];
    __syncthreads();
  }
  for (int c = 0; c < DSTATE; ++c) A32[j * DSTATE + c] = Ms[j * DSTATE + c];
}

// Phase 1: zero-init local scans per chunk (parallel over B*NCHUNK blocks).
__global__ __launch_bounds__(64) void scan_local_kernel(
    const float* __restrict__ u, float* __restrict__ Hloc,
    const float* __restrict__ A) {
  const int b = blockIdx.x / NCHUNK;
  const int c = blockIdx.x % NCHUNK;
  const int j = threadIdx.x;
  float arow[DSTATE];
#pragma unroll 8
  for (int k = 0; k < DSTATE; ++k) arow[k] = A[j * DSTATE + k];
  __shared__ float h[2][DSTATE];
  h[0][j] = 0.f;
  __syncthreads();
  int cur = 0;
  const size_t base = ((size_t)b * SEQ + (size_t)c * CHUNK) * DSTATE;
  for (int t = 0; t < CHUNK; ++t) {
    float acc = u[base + (size_t)t * DSTATE + j];
#pragma unroll 8
    for (int k = 0; k < DSTATE; ++k) acc += arow[k] * h[cur][k];
    h[cur ^ 1][j] = acc;
    Hloc[base + (size_t)t * DSTATE + j] = acc;
    __syncthreads();
    cur ^= 1;
  }
}

// Phase 2: sequential chunk-state combine, s_{c+1} = A32 s_c + local_end_c.
__global__ __launch_bounds__(64) void scan_combine_kernel(
    const float* __restrict__ Hloc, const float* __restrict__ A32,
    float* __restrict__ Sbuf) {
  const int b = blockIdx.x;
  const int j = threadIdx.x;
  float arow[DSTATE];
#pragma unroll 8
  for (int k = 0; k < DSTATE; ++k) arow[k] = A32[j * DSTATE + k];
  __shared__ float s[2][DSTATE];
  s[0][j] = 0.f;
  __syncthreads();
  int cur = 0;
  for (int c = 0; c < NCHUNK; ++c) {
    Sbuf[((size_t)b * NCHUNK + c) * DSTATE + j] = s[cur][j];
    float acc = Hloc[((size_t)b * SEQ + (size_t)c * CHUNK + (CHUNK - 1)) * DSTATE + j];
#pragma unroll 8
    for (int k = 0; k < DSTATE; ++k) acc += arow[k] * s[cur][k];
    s[cur ^ 1][j] = acc;
    __syncthreads();
    cur ^= 1;
  }
}

// Phase 3: fixup h_t = local_t + A^{i+1} s_chunk; write bf16 H for Cm GEMM.
__global__ __launch_bounds__(64) void scan_fixup_kernel(
    const float* __restrict__ Hloc, const float* __restrict__ Sbuf,
    const float* __restrict__ A, unsigned short* __restrict__ Hbf) {
  const int b = blockIdx.x / NCHUNK;
  const int c = blockIdx.x % NCHUNK;
  const int j = threadIdx.x;
  float arow[DSTATE];
#pragma unroll 8
  for (int k = 0; k < DSTATE; ++k) arow[k] = A[j * DSTATE + k];
  __shared__ float p[2][DSTATE];
  p[0][j] = Sbuf[((size_t)b * NCHUNK + c) * DSTATE + j];
  __syncthreads();
  int cur = 0;
  const size_t base = ((size_t)b * SEQ + (size_t)c * CHUNK) * DSTATE;
  for (int t = 0; t < CHUNK; ++t) {
    float acc = 0.f;
#pragma unroll 8
    for (int k = 0; k < DSTATE; ++k) acc += arow[k] * p[cur][k];
    p[cur ^ 1][j] = acc;
    const float h = Hloc[base + (size_t)t * DSTATE + j] + acc;
    Hbf[base + (size_t)t * DSTATE + j] = f2bf(h);
    __syncthreads();
    cur ^= 1;
  }
}

// ---------------------------------------------------------------------------
// Host-side orchestration
// ---------------------------------------------------------------------------
static inline size_t align256(size_t x) { return (x + 255) & ~(size_t)255; }

extern "C" void kernel_launch(void* const* d_in, const int* in_sizes, int n_in,
                              void* d_out, int out_size, void* d_ws, size_t ws_size,
                              hipStream_t stream) {
  (void)in_sizes; (void)n_in; (void)out_size; (void)ws_size;
  const float* x      = (const float*)d_in[0];
  const float* w1_w   = (const float*)d_in[1];
  const float* w1_b   = (const float*)d_in[2];
  const float* v1_w   = (const float*)d_in[3];
  const float* v1_b   = (const float*)d_in[4];
  const float* w2_w   = (const float*)d_in[5];
  const float* w2_b   = (const float*)d_in[6];
  const float* conv_w = (const float*)d_in[7];
  const float* Amat   = (const float*)d_in[8];
  const float* Bm     = (const float*)d_in[9];
  const float* Cm     = (const float*)d_in[10];
  const float* ln1_g  = (const float*)d_in[11];
  const float* ln1_b  = (const float*)d_in[12];
  const float* ln2_g  = (const float*)d_in[13];
  const float* ln2_b  = (const float*)d_in[14];
  const float* ff1_w  = (const float*)d_in[15];
  const float* ff1_b  = (const float*)d_in[16];
  const float* ff2_w  = (const float*)d_in[17];
  const float* ff2_b  = (const float*)d_in[18];
  float* out = (float*)d_out;

  char* ws = (char*)d_ws;
  size_t off = 0;
  auto carve = [&](size_t bytes) { size_t o = off; off = align256(off + bytes); return o; };

  const int M = MROWS, D = DMODEL, F = DFF, N = DSTATE;

  unsigned short* w1bf  = (unsigned short*)(ws + carve((size_t)D * D * 2));
  unsigned short* v1bf  = (unsigned short*)(ws + carve((size_t)D * D * 2));
  unsigned short* w2bf  = (unsigned short*)(ws + carve((size_t)D * D * 2));
  unsigned short* ff1bf = (unsigned short*)(ws + carve((size_t)F * D * 2));
  unsigned short* ff2bf = (unsigned short*)(ws + carve((size_t)D * F * 2));
  unsigned short* Bmbf  = (unsigned short*)(ws + carve((size_t)N * D * 2));
  unsigned short* Cmbf  = (unsigned short*)(ws + carve((size_t)D * N * 2));
  float*          A32   = (float*)(ws + carve((size_t)N * N * 4));
  unsigned short* h1bf  = (unsigned short*)(ws + carve((size_t)M * D * 2)); // also h2
  float*          bufA  = (float*)(ws + carve((size_t)M * D * 4));          // x1, then v
  unsigned short* xcbf  = (unsigned short*)(ws + carve((size_t)M * D * 2)); // also gate
  float*          u     = (float*)(ws + carve((size_t)M * N * 4));
  float*          Hloc  = (float*)(ws + carve((size_t)M * N * 4));
  float*          Sbuf  = (float*)(ws + carve((size_t)BATCH * NCHUNK * N * 4));
  unsigned short* Hbf   = (unsigned short*)(ws + carve((size_t)M * N * 2));
  float*          xssm  = (float*)(ws + carve((size_t)M * D * 4));          // also ff1out bf16
  float*          x2    = (float*)(ws + carve((size_t)M * D * 4));
  unsigned short* ff1o  = (unsigned short*)xssm;   // reuse (consumed before)
  unsigned short* gbf   = xcbf;                    // reuse (consumed before)
  unsigned short* h2bf  = h1bf;                    // reuse (consumed before)

  auto cvt = [&](const float* s, unsigned short* d, long long n) {
    int blocks = (int)((n + 255) / 256);
    if (blocks > 16384) blocks = 16384;
    f32_to_bf16_kernel<<<blocks, 256, 0, stream>>>(s, d, n);
  };

  // --- weight conversions (f32 -> bf16) ------------------------------------
  cvt(w1_w, w1bf, (long long)D * D);
  cvt(v1_w, v1bf, (long long)D * D);
  cvt(w2_w, w2bf, (long long)D * D);
  cvt(ff1_w, ff1bf, (long long)F * D);
  cvt(ff2_w, ff2bf, (long long)D * F);
  cvt(Bm, Bmbf, (long long)N * D);
  cvt(Cm, Cmbf, (long long)D * N);
  apow32_kernel<<<1, 64, 0, stream>>>(Amat, A32);

  // --- LN1 -> h1 (bf16) -----------------------------------------------------
  layernorm_bf16_kernel<<<M, 256, 0, stream>>>(x, ln1_g, ln1_b, h1bf, D);

  // --- x1 = h1 @ w1^T + b ---------------------------------------------------
  gemm_bf16_kernel<EPI_NONE, false, 2><<<dim3(D / 128, M / 128), 256, 0, stream>>>(
      h1bf, w1bf, w1_b, nullptr, bufA, M, D, D);

  // --- xc = silu(depthwise_conv(x1)) (bf16) --------------------------------
  conv_silu_kernel<<<(int)(((long long)M * D + 255) / 256), 256, 0, stream>>>(
      bufA, conv_w, xcbf, SEQ, D, (long long)M * D);

  // --- v = silu(h1 @ v1^T + b)  (f32, reuses bufA after conv consumed x1) --
  gemm_bf16_kernel<EPI_SILU, false, 2><<<dim3(D / 128, M / 128), 256, 0, stream>>>(
      h1bf, v1bf, v1_b, nullptr, bufA, M, D, D);

  // --- u = xc @ Bm^T  [M,64]  (N=64 -> NT=1 tile shape) --------------------
  gemm_bf16_kernel<EPI_NONE, false, 1><<<dim3(N / 64, M / 128), 256, 0, stream>>>(
      xcbf, Bmbf, nullptr, nullptr, u, M, N, D);

  // --- chunked parallel scan -> Hbf ----------------------------------------
  scan_local_kernel<<<BATCH * NCHUNK, 64, 0, stream>>>(u, Hloc, Amat);
  scan_combine_kernel<<<BATCH, 64, 0, stream>>>(Hloc, A32, Sbuf);
  scan_fixup_kernel<<<BATCH * NCHUNK, 64, 0, stream>>>(Hloc, Sbuf, Amat, Hbf);

  // --- x_ssm = H @ Cm^T  [M,1024], K=64 ------------------------------------
  gemm_bf16_kernel<EPI_NONE, false, 2><<<dim3(D / 128, M / 128), 256, 0, stream>>>(
      Hbf, Cmbf, nullptr, nullptr, xssm, M, D, N);

  // --- gate = bf16(x_ssm * v) ----------------------------------------------
  gate_bf16_kernel<<<(int)(((long long)M * D + 255) / 256), 256, 0, stream>>>(
      xssm, bufA, gbf, (long long)M * D);

  // --- x2 = gate @ w2^T + b + x  (first residual) --------------------------
  gemm_bf16_kernel<EPI_RESID, false, 2><<<dim3(D / 128, M / 128), 256, 0, stream>>>(
      gbf, w2bf, w2_b, x, x2, M, D, D);

  // --- LN2 -> h2 (bf16) -----------------------------------------------------
  layernorm_bf16_kernel<<<M, 256, 0, stream>>>(x2, ln2_g, ln2_b, h2bf, D);

  // --- ff1: gelu(h2 @ ff1^T + b) -> bf16 [M,2048] ---------------------------
  gemm_bf16_kernel<EPI_GELU, true, 2><<<dim3(F / 128, M / 128), 256, 0, stream>>>(
      h2bf, ff1bf, ff1_b, nullptr, ff1o, M, F, D);

  // --- out = ff1o @ ff2^T + b + x2  (second residual) ----------------------
  gemm_bf16_kernel<EPI_RESID, false, 2><<<dim3(D / 128, M / 128), 256, 0, stream>>>(
      ff1o, ff2bf, ff2_b, x2, out, M, D, F);
}
